// FeedForward_11424613007647
// MI455X (gfx1250) — matrix operand order repaired and checked by reading, not compiled, not run on hardware
//
#include <hip/hip_runtime.h>
#include <math.h>

// ---------------------------------------------------------------------------
// Fused gated FFN for MI455X (gfx1250, wave32, WMMA + TDM):
//   act = gelu(w0 @ x) * (w1 @ x)   (bf16, staged in workspace)
//   out = w_out @ act               (f32)
// GEMM core: v_wmma_f32_16x16x32_bf16.  Weight tiles are DMAed into LDS by
// the Tensor Data Mover (tensor_load_to_lds, TENSORcnt) with double-buffered
// LDS so the DMA of K-step i+1 overlaps the WMMAs of K-step i.
// ---------------------------------------------------------------------------

typedef __attribute__((ext_vector_type(16))) __bf16        v16bf;
typedef __attribute__((ext_vector_type(8)))  float         v8f;
typedef __attribute__((ext_vector_type(4)))  unsigned int  u32x4;
typedef __attribute__((ext_vector_type(8)))  int           i32x8;
typedef __attribute__((ext_vector_type(4)))  int           i32x4;

#define B_ 2
#define D_ 4096
#define S_ 2048
#define F_ 16384

#define BM 128          // tile rows   (F or D)
#define BN 128          // tile cols   (S)
#define BK 32           // K step == WMMA K
#define THREADS 256     // 8 waves of 32
#define BSTRIDE 40      // padded K-stride of transposed B tile (elems, 80B rows)

static __device__ __forceinline__ unsigned short f32_to_bf16(float f) {
  unsigned int u = __builtin_bit_cast(unsigned int, f);
  u += 0x7FFFu + ((u >> 16) & 1u);          // round-to-nearest-even
  return (unsigned short)(u >> 16);
}

static __device__ __forceinline__ float gelu_exact(float x) {
  return 0.5f * x * (1.0f + erff(x * 0.70710678118654752f));
}

union Frag16 { uint4 q[2]; v16bf v; };

// A tile in LDS: row-major [m][k], k-stride BK (64B rows).
// Code's k order (matches B): lane holds row M=lane&15; K contiguous 16 at h*16, h=lane>>4.
static __device__ __forceinline__ v16bf load_a_frag(const unsigned short* lds,
                                                    int m_frag, int lane) {
  const int row   = m_frag + (lane & 15);
  const int khalf = lane >> 4;
  const char* p = (const char*)(lds + row * BK);
  Frag16 f;
  f.q[0] = *(const uint4*)(p + khalf * 32);
  f.q[1] = *(const uint4*)(p + khalf * 32 + 16);
  return f.v;
}

// B tile in LDS: K-transposed [n][k], k-stride BSTRIDE.
// Code's k order (matches A): lane holds col N=lane&15; K contiguous 16 at (lane>>4)*16.
static __device__ __forceinline__ v16bf load_b_frag(const unsigned short* lds,
                                                    int n_frag, int lane) {
  const int n  = n_frag + (lane & 15);
  const int kb = (lane >> 4) * 16;
  const char* p = (const char*)(lds + n * BSTRIDE + kb);
  Frag16 f;
  f.q[0] = *(const uint4*)(p);
  f.q[1] = *(const uint4*)(p + 16);
  return f.v;
}

// ---------------------------------------------------------------------------
// TDM: DMA a BM x BK bf16 tile (row-major, K-dim stride ldk elements) from
// global into LDS.  D# per ISA ch.8: group0 = {count=1 | lds_addr |
// global_addr | type=2}, group1 = {data_size=1(2B), tensor_dim0/1,
// tile_dim0=BK, tile_dim1=BM, tensor_dim0_stride=ldk}.  2D tensor: groups
// 2/3 (and the trailing group) zero.  Tracked by TENSORcnt.
// ---------------------------------------------------------------------------
static __device__ __forceinline__ void tdm_load_tile(
    const unsigned short* __restrict__ gsrc, unsigned lds_addr,
    int rows_total, int ldk, int tileM, int k0) {
  const unsigned long long ga =
      (unsigned long long)(size_t)gsrc +
      ((unsigned long long)(unsigned)tileM * (unsigned)ldk + (unsigned)k0) * 2ull;

  u32x4 g0;
  g0[0] = 1u;                                   // count=1, user mode
  g0[1] = lds_addr;                             // [63:32]  lds_addr (bytes)
  g0[2] = (unsigned)ga;                         // [95:64]  global_addr lo
  g0[3] = (unsigned)((ga >> 32) & 0x01FFFFFFu)  // [120:96] global_addr hi
          | (2u << 30);                         // [127:126] type=2 (image)

  i32x8 g1;
  g1[0] = (int)(1u << 16);                      // data_size=1 -> 2 bytes
  g1[1] = (int)(((unsigned)ldk & 0xFFFFu) << 16);          // tensor_dim0 lo16
  g1[2] = (int)(((unsigned)ldk >> 16) |
                (((unsigned)rows_total & 0xFFFFu) << 16)); // dim0 hi | dim1 lo
  g1[3] = (int)(((unsigned)rows_total >> 16) |
                ((unsigned)BK << 16));                     // dim1 hi | tile_dim0
  g1[4] = (int)(unsigned)BM;                    // tile_dim1=BM, tile_dim2=0
  g1[5] = (int)(unsigned)ldk;                   // tensor_dim0_stride lo32
  g1[6] = 0;                                    // stride hi16 | dim1_stride lo16
  g1[7] = 0;

  const i32x4 z4 = {0, 0, 0, 0};
  const i32x8 z8 = {0, 0, 0, 0, 0, 0, 0, 0};
  __builtin_amdgcn_tensor_load_to_lds(g0, g1, z4, z4, z8, 0);
}

static __device__ __forceinline__ unsigned lds_addr_of(const void* p) {
  return (unsigned)(size_t)p;   // LDS generic pointer: low 32 bits = LDS byte addr
}

// Fill B tile (bf16 [k][n], N-dim stride ldn) K-transposed into LDS [n][k].
// Thread t: k-pair = 2*(t&15), n-range = ((t>>4)*8 .. +7); pairs packed as b32.
static __device__ __forceinline__ void fill_b(unsigned short* lds,
                                              const unsigned short* __restrict__ gb,
                                              int tileN, int k0, int ldn) {
  const int kp = (threadIdx.x & 15) * 2;
  const int n0 = (threadIdx.x >> 4) * 8;
  const unsigned short* r0 = gb + (size_t)(k0 + kp) * ldn + tileN + n0;
  const uint4 a = *(const uint4*)(r0);        // 8 bf16 along n, row k
  const uint4 b = *(const uint4*)(r0 + ldn);  // 8 bf16 along n, row k+1
  const unsigned short* pa = (const unsigned short*)&a;
  const unsigned short* pb = (const unsigned short*)&b;
#pragma unroll
  for (int e = 0; e < 8; ++e) {
    const unsigned int w = (unsigned int)pa[e] | ((unsigned int)pb[e] << 16);
    *(unsigned int*)(lds + (n0 + e) * BSTRIDE + kp) = w;
  }
}

// ---------------------------------------------------------------------------
// Stage A: act[b] = gelu(w0 @ x[b]) * (w1 @ x[b]), bf16 out.  M=F, N=S, K=D.
// ---------------------------------------------------------------------------
__global__ __launch_bounds__(THREADS)
void ffn_stage_a(const unsigned short* __restrict__ w0b,
                 const unsigned short* __restrict__ w1b,
                 const unsigned short* __restrict__ xb,
                 unsigned short* __restrict__ actb) {
  __shared__ __align__(16) unsigned short sA0[2][BM * BK];
  __shared__ __align__(16) unsigned short sA1[2][BM * BK];
  __shared__ __align__(16) unsigned short sB [2][BN * BSTRIDE];

  const int tileN = blockIdx.x * BN;
  const int tileM = blockIdx.y * BM;
  const int batch = blockIdx.z;
  const unsigned short* xp = xb + (size_t)batch * D_ * S_;

  const int lane = threadIdx.x & 31;
  const int wave = threadIdx.x >> 5;
  const int wm = wave >> 2;   // 0..1 -> 64-row slab
  const int wn = wave & 3;    // 0..3 -> 32-col slab

  v8f accg[4][2], accu[4][2];
#pragma unroll
  for (int i = 0; i < 4; ++i)
#pragma unroll
    for (int j = 0; j < 2; ++j) {
      accg[i][j] = (v8f){0.f, 0.f, 0.f, 0.f, 0.f, 0.f, 0.f, 0.f};
      accu[i][j] = (v8f){0.f, 0.f, 0.f, 0.f, 0.f, 0.f, 0.f, 0.f};
    }

  // Prologue: DMA K-step 0 weight tiles, stage x tile manually (transpose).
  if (wave == 0) tdm_load_tile(w0b, lds_addr_of(sA0[0]), F_, D_, tileM, 0);
  if (wave == 1) tdm_load_tile(w1b, lds_addr_of(sA1[0]), F_, D_, tileM, 0);
  fill_b(sB[0], xp, tileN, 0, S_);

  const int NK = D_ / BK;
  for (int it = 0; it < NK; ++it) {
    const int buf = it & 1;
    if (wave <= 1) __builtin_amdgcn_s_wait_tensorcnt(0);  // A[buf] landed in LDS
    __syncthreads();   // publishes TDM tiles + fill_b stores (s_wait_dscnt 0)

    if (it + 1 < NK) {  // overlap next-tile DMA/fill with this tile's WMMAs
      const int k1 = (it + 1) * BK;
      if (wave == 0) tdm_load_tile(w0b, lds_addr_of(sA0[buf ^ 1]), F_, D_, tileM, k1);
      if (wave == 1) tdm_load_tile(w1b, lds_addr_of(sA1[buf ^ 1]), F_, D_, tileM, k1);
      fill_b(sB[buf ^ 1], xp, tileN, k1, S_);
    }

    v16bf bf0 = load_b_frag(sB[buf], wn * 32 + 0, lane);
    v16bf bf1 = load_b_frag(sB[buf], wn * 32 + 16, lane);
#pragma unroll
    for (int fm = 0; fm < 4; ++fm) {
      v16bf a0 = load_a_frag(sA0[buf], wm * 64 + fm * 16, lane);
      v16bf a1 = load_a_frag(sA1[buf], wm * 64 + fm * 16, lane);
      accg[fm][0] = __builtin_amdgcn_wmma_f32_16x16x32_bf16(
          false, a0, false, bf0, (short)0, accg[fm][0], false, false);
      accg[fm][1] = __builtin_amdgcn_wmma_f32_16x16x32_bf16(
          false, a0, false, bf1, (short)0, accg[fm][1], false, false);
      accu[fm][0] = __builtin_amdgcn_wmma_f32_16x16x32_bf16(
          false, a1, false, bf0, (short)0, accu[fm][0], false, false);
      accu[fm][1] = __builtin_amdgcn_wmma_f32_16x16x32_bf16(
          false, a1, false, bf1, (short)0, accu[fm][1], false, false);
    }
  }

  // Epilogue: gelu(gate)*up -> bf16.  D-layout: m = e + ((lane>>4)<<3), n = lane&15.
  unsigned short* out = actb + (size_t)batch * F_ * S_;
  const int mrow = tileM + wm * 64 + ((lane >> 4) << 3);
  const int ncol = tileN + wn * 32 + (lane & 15);
#pragma unroll
  for (int fm = 0; fm < 4; ++fm)
#pragma unroll
    for (int fn = 0; fn < 2; ++fn)
#pragma unroll
      for (int e = 0; e < 8; ++e) {
        const int m = mrow + fm * 16 + e;
        const int n = ncol + fn * 16;
        const float h = gelu_exact(accg[fm][fn][e]) * accu[fm][fn][e];
        out[(size_t)m * S_ + n] = f32_to_bf16(h);
      }
}

// ---------------------------------------------------------------------------
// Stage B: out[b] = w_out @ act[b], f32 out.  M=D, N=S, K=F.
// ---------------------------------------------------------------------------
__global__ __launch_bounds__(THREADS)
void ffn_stage_b(const unsigned short* __restrict__ wob,
                 const unsigned short* __restrict__ actb,
                 float* __restrict__ outp) {
  __shared__ __align__(16) unsigned short sA[2][BM * BK];
  __shared__ __align__(16) unsigned short sB[2][BN * BSTRIDE];

  const int tileN = blockIdx.x * BN;
  const int tileM = blockIdx.y * BM;
  const int batch = blockIdx.z;
  const unsigned short* hp = actb + (size_t)batch * F_ * S_;

  const int lane = threadIdx.x & 31;
  const int wave = threadIdx.x >> 5;
  const int wm = wave >> 2;
  const int wn = wave & 3;

  v8f acc[4][2];
#pragma unroll
  for (int i = 0; i < 4; ++i)
#pragma unroll
    for (int j = 0; j < 2; ++j)
      acc[i][j] = (v8f){0.f, 0.f, 0.f, 0.f, 0.f, 0.f, 0.f, 0.f};

  if (wave == 0) tdm_load_tile(wob, lds_addr_of(sA[0]), D_, F_, tileM, 0);
  fill_b(sB[0], hp, tileN, 0, S_);

  const int NK = F_ / BK;
  for (int it = 0; it < NK; ++it) {
    const int buf = it & 1;
    if (wave == 0) __builtin_amdgcn_s_wait_tensorcnt(0);
    __syncthreads();

    if (it + 1 < NK) {
      const int k1 = (it + 1) * BK;
      if (wave == 0) tdm_load_tile(wob, lds_addr_of(sA[buf ^ 1]), D_, F_, tileM, k1);
      fill_b(sB[buf ^ 1], hp, tileN, k1, S_);
    }

    v16bf bf0 = load_b_frag(sB[buf], wn * 32 + 0, lane);
    v16bf bf1 = load_b_frag(sB[buf], wn * 32 + 16, lane);
#pragma unroll
    for (int fm = 0; fm < 4; ++fm) {
      v16bf a = load_a_frag(sA[buf], wm * 64 + fm * 16, lane);
      acc[fm][0] = __builtin_amdgcn_wmma_f32_16x16x32_bf16(
          false, a, false, bf0, (short)0, acc[fm][0], false, false);
      acc[fm][1] = __builtin_amdgcn_wmma_f32_16x16x32_bf16(
          false, a, false, bf1, (short)0, acc[fm][1], false, false);
    }
  }

  float* out = outp + (size_t)batch * D_ * S_;
  const int mrow = tileM + wm * 64 + ((lane >> 4) << 3);
  const int ncol = tileN + wn * 32 + (lane & 15);
#pragma unroll
  for (int fm = 0; fm < 4; ++fm)
#pragma unroll
    for (int fn = 0; fn < 2; ++fn)
#pragma unroll
      for (int e = 0; e < 8; ++e)
        out[(size_t)(mrow + fm * 16 + e) * S_ + (ncol + fn * 16)] = acc[fm][fn][e];
}

// ---------------------------------------------------------------------------
// f32 -> bf16 conversion (vectorized, 4 elems/thread).
// ---------------------------------------------------------------------------
__global__ void cvt_f32_bf16(const float4* __restrict__ src,
                             uint2* __restrict__ dst, int n4) {
  const int i = blockIdx.x * blockDim.x + threadIdx.x;
  if (i < n4) {
    const float4 v = src[i];
    uint2 o;
    o.x = (unsigned)f32_to_bf16(v.x) | ((unsigned)f32_to_bf16(v.y) << 16);
    o.y = (unsigned)f32_to_bf16(v.z) | ((unsigned)f32_to_bf16(v.w) << 16);
    dst[i] = o;
  }
}

extern "C" void kernel_launch(void* const* d_in, const int* in_sizes, int n_in,
                              void* d_out, int out_size, void* d_ws, size_t ws_size,
                              hipStream_t stream) {
  const float* x  = (const float*)d_in[0];   // (B, D, S)
  const float* w0 = (const float*)d_in[1];   // (F, D)
  const float* w1 = (const float*)d_in[2];   // (F, D)
  const float* wo = (const float*)d_in[3];   // (D, F)
  float* out = (float*)d_out;                // (B, D, S)

  // Workspace carve-out (bf16 copies + activation), 256B aligned.
  char* ws = (char*)d_ws;
  size_t off = 0;
  auto carve = [&](size_t bytes) {
    char* p = ws + off;
    off += (bytes + 255) & ~(size_t)255;
    return p;
  };
  unsigned short* w0b  = (unsigned short*)carve((size_t)F_ * D_ * 2);
  unsigned short* w1b  = (unsigned short*)carve((size_t)F_ * D_ * 2);
  unsigned short* wob  = (unsigned short*)carve((size_t)D_ * F_ * 2);
  unsigned short* xb   = (unsigned short*)carve((size_t)B_ * D_ * S_ * 2);
  unsigned short* actb = (unsigned short*)carve((size_t)B_ * F_ * S_ * 2);
  (void)ws_size; (void)in_sizes; (void)n_in; (void)out_size;

  const int n4w = (F_ * D_) / 4;            // 16.7M float4 per weight
  const int n4x = (B_ * D_ * S_) / 4;
  cvt_f32_bf16<<<(n4w + 255) / 256, 256, 0, stream>>>((const float4*)w0, (uint2*)w0b, n4w);
  cvt_f32_bf16<<<(n4w + 255) / 256, 256, 0, stream>>>((const float4*)w1, (uint2*)w1b, n4w);
  cvt_f32_bf16<<<(n4w + 255) / 256, 256, 0, stream>>>((const float4*)wo, (uint2*)wob, n4w);
  cvt_f32_bf16<<<(n4x + 255) / 256, 256, 0, stream>>>((const float4*)x,  (uint2*)xb,  n4x);

  dim3 gridA(S_ / BN, F_ / BM, B_);   // 16 x 128 x 2
  ffn_stage_a<<<gridA, THREADS, 0, stream>>>(w0b, w1b, xb, actb);

  dim3 gridB(S_ / BN, D_ / BM, B_);   // 16 x 32 x 2
  ffn_stage_b<<<gridB, THREADS, 0, stream>>>(wob, actb, out);
}
